// GRUModule_90701119357081
// MI455X (gfx1250) — matrix operand order repaired and checked by reading, not compiled, not run on hardware
//
#include <hip/hip_runtime.h>
#include <hip/hip_bf16.h>

#define BATCH 8192
#define IN_DIM 1024
#define HID 2048
#define OUT_DIM 1024

#define BM 64
#define BN 128
#define BK 32

typedef __bf16  bf16x16 __attribute__((ext_vector_type(16)));
typedef __bf16  bf16x8  __attribute__((ext_vector_type(8)));
typedef float   f32x8   __attribute__((ext_vector_type(8)));
typedef unsigned long long u64;

#define WMMA_BF16(a, b, c) \
    __builtin_amdgcn_wmma_f32_16x16x32_bf16(false, (a), false, (b), (short)0, (c), false, false)

// ---------------- async global->LDS path (gfx1250) with sync fallback ------
#if __has_builtin(__builtin_amdgcn_global_load_async_to_lds_b128)
#define USE_ASYNC 1
#else
#define USE_ASYNC 0
#endif

#if USE_ASYNC
// Builtin signature (from hipcc diagnostic): first param is
// '<4 x i32> addrspace(1)*' (non-const). Mirror that for the LDS side (AS3).
typedef int i32x4v __attribute__((vector_size(16)));
typedef __attribute__((address_space(1))) i32x4v* gas_ptr;
typedef __attribute__((address_space(3))) i32x4v* las_ptr;
__device__ __forceinline__ void async_b128(const __bf16* g, __bf16* l) {
    __builtin_amdgcn_global_load_async_to_lds_b128((gas_ptr)g, (las_ptr)l, 0, 0);
}
template <int N>
__device__ __forceinline__ void wait_async() {
#if __has_builtin(__builtin_amdgcn_s_wait_asynccnt)
    __builtin_amdgcn_s_wait_asynccnt(N);
#else
    asm volatile("s_wait_asynccnt %0" ::"i"(N) : "memory");
#endif
}
#else
template <int N>
__device__ __forceinline__ void wait_async() {}
#endif

__device__ __forceinline__ float sigmoidf_(float x) {
    return 1.0f / (1.0f + __expf(-x));
}

// ---------------- f32 -> bf16 cast ----------------
__global__ void cast_f32_bf16(const float* __restrict__ s, __bf16* __restrict__ d, long n) {
    long i = (long)blockIdx.x * blockDim.x + threadIdx.x;
    long stride = (long)gridDim.x * blockDim.x;
    for (; i < n; i += stride) d[i] = (__bf16)s[i];
}

// ---------------- tile copy: global -> LDS ----------------
// A tile: BM x BK bf16, k contiguous. 64*32*2B = 4KB = 256 x 16B chunks.
__device__ __forceinline__ void copy_tileA(__bf16* sA, const __bf16* gA, int lda,
                                           int rowbase, int kbase, int tid) {
#if USE_ASYNC
    const int row = tid >> 2;
    const int c   = (tid & 3) * 8;
    async_b128(gA + (size_t)(rowbase + row) * lda + kbase + c, sA + row * BK + c);
#else
    const u64* g = (const u64*)gA;
    u64* s = (u64*)sA;
#pragma unroll
    for (int i = 0; i < 2; i++) {
        int idx = tid + i * 256;  // 0..511 u64 chunks
        int row = idx >> 3;
        int c   = idx & 7;
        s[idx] = g[(size_t)(rowbase + row) * (size_t)(lda >> 2) + (size_t)(kbase >> 2) + c];
    }
#endif
}

// W tile: BN x BK bf16, stored [n][k]. 128*32*2B = 8KB = 512 x 16B chunks.
__device__ __forceinline__ void copy_tileW(__bf16* sW, const __bf16* gW, int ldw,
                                           int nbase, int kbase, int tid) {
#if USE_ASYNC
#pragma unroll
    for (int i = 0; i < 2; i++) {
        int idx = tid + i * 256;  // 0..511
        int row = idx >> 2;
        int c   = (idx & 3) * 8;
        async_b128(gW + (size_t)(nbase + row) * ldw + kbase + c, sW + row * BK + c);
    }
#else
    const u64* g = (const u64*)gW;
    u64* s = (u64*)sW;
#pragma unroll
    for (int i = 0; i < 4; i++) {
        int idx = tid + i * 256;  // 0..1023 u64 chunks
        int row = idx >> 3;
        int c   = idx & 7;
        s[idx] = g[(size_t)(nbase + row) * (size_t)(ldw >> 2) + (size_t)(kbase >> 2) + c];
    }
#endif
}

// ---------------- fragment loads (contiguous-k layouts, same for A and B) ----------------
// A frag (16-bit A, 16x32): lane half h, r = lane&15 -> row M=r;
//   elems 0..15 -> K = h*16 + e
__device__ __forceinline__ bf16x16 load_afrag(const __bf16* sA, int mrow, int lane) {
    const int half = lane >> 4, r = lane & 15;
    const __bf16* base = sA + (size_t)(mrow + r) * BK + half * 16;
    bf16x8 lo = *(const bf16x8*)(base);
    bf16x8 hi = *(const bf16x8*)(base + 8);
    bf16x16 t;
#pragma unroll
    for (int e = 0; e < 8; e++) { t[e] = lo[e]; t[e + 8] = hi[e]; }
    return t;
}

// B frag (K x N = W^T): lane half h, col N=r; elems 0..15 -> K = h*16 + e.
// B[k][n] = W[n][k] -> sW[n*BK + h*16 + e] contiguous.
__device__ __forceinline__ bf16x16 load_bfrag(const __bf16* sW, int ncol, int lane) {
    const int half = lane >> 4, r = lane & 15;
    return *(const bf16x16*)(sW + (size_t)(ncol + r) * BK + half * 16);
}

// ---------------- Kernel 1: r and z gates (fused dual GEMM, pipelined) -----
__global__ __launch_bounds__(256) void gemm_rz(
    const __bf16* __restrict__ xb,  const __bf16* __restrict__ hb,
    const __bf16* __restrict__ wxr, const __bf16* __restrict__ wxz,
    const __bf16* __restrict__ whr, const __bf16* __restrict__ whz,
    const float* __restrict__ bxr,  const float* __restrict__ bxz,
    const float* __restrict__ hidden_f32,
    __bf16* __restrict__ rh_out, float* __restrict__ z_out)
{
    __shared__ __align__(16) __bf16 sA[2][BM * BK];
    __shared__ __align__(16) __bf16 sW0[2][BN * BK];
    __shared__ __align__(16) __bf16 sW1[2][BN * BK];

    const int tid  = threadIdx.x;
    const int lane = tid & 31;
    const int w    = tid >> 5;
    const int wm   = w & 1;    // 2 waves along M
    const int wn   = w >> 1;   // 4 waves along N
    const int mbase = blockIdx.y * BM;
    const int nbase = blockIdx.x * BN;

    f32x8 accr[2][2] = {};
    f32x8 accz[2][2] = {};

    constexpr int SX    = IN_DIM / BK;        // 32
    constexpr int STEPS = SX + HID / BK;      // 96

    auto issue = [&](int s, int buf) {
        const __bf16 *A, *W0, *W1;
        int ld, k;
        if (s < SX) { A = xb; W0 = wxr; W1 = wxz; ld = IN_DIM; k = s * BK; }
        else        { A = hb; W0 = whr; W1 = whz; ld = HID;    k = (s - SX) * BK; }
        copy_tileA(sA[buf], A, ld, mbase, k, tid);
        copy_tileW(sW0[buf], W0, ld, nbase, k, tid);
        copy_tileW(sW1[buf], W1, ld, nbase, k, tid);
    };

    issue(0, 0);
    for (int s = 0; s < STEPS; s++) {
        const int cur = s & 1;
        if (s + 1 < STEPS) { issue(s + 1, cur ^ 1); wait_async<5>(); }
        else               { wait_async<0>(); }
        __syncthreads();

        bf16x16 a0 = load_afrag(sA[cur], wm * 32,      lane);
        bf16x16 a1 = load_afrag(sA[cur], wm * 32 + 16, lane);
        bf16x16 br0 = load_bfrag(sW0[cur], wn * 32,      lane);
        bf16x16 br1 = load_bfrag(sW0[cur], wn * 32 + 16, lane);
        bf16x16 bz0 = load_bfrag(sW1[cur], wn * 32,      lane);
        bf16x16 bz1 = load_bfrag(sW1[cur], wn * 32 + 16, lane);

        accr[0][0] = WMMA_BF16(a0, br0, accr[0][0]);
        accr[0][1] = WMMA_BF16(a0, br1, accr[0][1]);
        accr[1][0] = WMMA_BF16(a1, br0, accr[1][0]);
        accr[1][1] = WMMA_BF16(a1, br1, accr[1][1]);
        accz[0][0] = WMMA_BF16(a0, bz0, accz[0][0]);
        accz[0][1] = WMMA_BF16(a0, bz1, accz[0][1]);
        accz[1][0] = WMMA_BF16(a1, bz0, accz[1][0]);
        accz[1][1] = WMMA_BF16(a1, bz1, accz[1][1]);

        __syncthreads();
    }

    const int half = lane >> 4;
    const int lr   = lane & 15;
#pragma unroll
    for (int i = 0; i < 2; i++)
#pragma unroll
        for (int j = 0; j < 2; j++) {
            const int colv = nbase + wn * 32 + j * 16 + lr;
            const float br = bxr[colv];
            const float bz = bxz[colv];
#pragma unroll
            for (int g = 0; g < 8; g++) {
                const int rowv = mbase + wm * 32 + i * 16 + g + half * 8;
                const size_t idx = (size_t)rowv * HID + colv;
                const float rv = sigmoidf_(accr[i][j][g] + br);
                const float zv = sigmoidf_(accz[i][j][g] + bz);
                const float hv = hidden_f32[idx];
                rh_out[idx] = (__bf16)(rv * hv);
                z_out[idx]  = zv;
            }
        }
}

// ---------------- Kernel 2: candidate + blend (pipelined) ----------------
__global__ __launch_bounds__(256) void gemm_h(
    const __bf16* __restrict__ xb,  const __bf16* __restrict__ rhb,
    const __bf16* __restrict__ wxh, const __bf16* __restrict__ whh,
    const float* __restrict__ bxh,
    const float* __restrict__ z_in, const float* __restrict__ hidden_f32,
    float* __restrict__ nh_f32, __bf16* __restrict__ nh_bf16)
{
    __shared__ __align__(16) __bf16 sA[2][BM * BK];
    __shared__ __align__(16) __bf16 sW[2][BN * BK];

    const int tid  = threadIdx.x;
    const int lane = tid & 31;
    const int w    = tid >> 5;
    const int wm   = w & 1;
    const int wn   = w >> 1;
    const int mbase = blockIdx.y * BM;
    const int nbase = blockIdx.x * BN;

    f32x8 acc[2][2] = {};

    constexpr int SX    = IN_DIM / BK;    // 32
    constexpr int STEPS = SX + HID / BK;  // 96

    auto issue = [&](int s, int buf) {
        const __bf16 *A, *W;
        int ld, k;
        if (s < SX) { A = xb;  W = wxh; ld = IN_DIM; k = s * BK; }
        else        { A = rhb; W = whh; ld = HID;    k = (s - SX) * BK; }
        copy_tileA(sA[buf], A, ld, mbase, k, tid);
        copy_tileW(sW[buf], W, ld, nbase, k, tid);
    };

    issue(0, 0);
    for (int s = 0; s < STEPS; s++) {
        const int cur = s & 1;
        if (s + 1 < STEPS) { issue(s + 1, cur ^ 1); wait_async<3>(); }
        else               { wait_async<0>(); }
        __syncthreads();

        bf16x16 a0 = load_afrag(sA[cur], wm * 32,      lane);
        bf16x16 a1 = load_afrag(sA[cur], wm * 32 + 16, lane);
        bf16x16 b0 = load_bfrag(sW[cur], wn * 32,      lane);
        bf16x16 b1 = load_bfrag(sW[cur], wn * 32 + 16, lane);

        acc[0][0] = WMMA_BF16(a0, b0, acc[0][0]);
        acc[0][1] = WMMA_BF16(a0, b1, acc[0][1]);
        acc[1][0] = WMMA_BF16(a1, b0, acc[1][0]);
        acc[1][1] = WMMA_BF16(a1, b1, acc[1][1]);

        __syncthreads();
    }

    const int half = lane >> 4;
    const int lr   = lane & 15;
#pragma unroll
    for (int i = 0; i < 2; i++)
#pragma unroll
        for (int j = 0; j < 2; j++) {
            const int colv = nbase + wn * 32 + j * 16 + lr;
            const float bh = bxh[colv];
#pragma unroll
            for (int g = 0; g < 8; g++) {
                const int rowv = mbase + wm * 32 + i * 16 + g + half * 8;
                const size_t idx = (size_t)rowv * HID + colv;
                const float hc = tanhf(acc[i][j][g] + bh);
                const float zv = z_in[idx];
                const float hv = hidden_f32[idx];
                const float nh = zv * hv + (1.0f - zv) * hc;
                nh_f32[idx]  = nh;
                nh_bf16[idx] = (__bf16)nh;
            }
        }
}

// ---------------- Kernel 3: output projection (pipelined) ----------------
__global__ __launch_bounds__(256) void gemm_out(
    const __bf16* __restrict__ nhb, const __bf16* __restrict__ who,
    const float* __restrict__ bo, float* __restrict__ out)
{
    __shared__ __align__(16) __bf16 sA[2][BM * BK];
    __shared__ __align__(16) __bf16 sW[2][BN * BK];

    const int tid  = threadIdx.x;
    const int lane = tid & 31;
    const int w    = tid >> 5;
    const int wm   = w & 1;
    const int wn   = w >> 1;
    const int mbase = blockIdx.y * BM;
    const int nbase = blockIdx.x * BN;

    f32x8 acc[2][2] = {};

    constexpr int STEPS = HID / BK;  // 64

    auto issue = [&](int s, int buf) {
        const int k = s * BK;
        copy_tileA(sA[buf], nhb, HID, mbase, k, tid);
        copy_tileW(sW[buf], who, HID, nbase, k, tid);
    };

    issue(0, 0);
    for (int s = 0; s < STEPS; s++) {
        const int cur = s & 1;
        if (s + 1 < STEPS) { issue(s + 1, cur ^ 1); wait_async<3>(); }
        else               { wait_async<0>(); }
        __syncthreads();

        bf16x16 a0 = load_afrag(sA[cur], wm * 32,      lane);
        bf16x16 a1 = load_afrag(sA[cur], wm * 32 + 16, lane);
        bf16x16 b0 = load_bfrag(sW[cur], wn * 32,      lane);
        bf16x16 b1 = load_bfrag(sW[cur], wn * 32 + 16, lane);

        acc[0][0] = WMMA_BF16(a0, b0, acc[0][0]);
        acc[0][1] = WMMA_BF16(a0, b1, acc[0][1]);
        acc[1][0] = WMMA_BF16(a1, b0, acc[1][0]);
        acc[1][1] = WMMA_BF16(a1, b1, acc[1][1]);

        __syncthreads();
    }

    const int half = lane >> 4;
    const int lr   = lane & 15;
#pragma unroll
    for (int i = 0; i < 2; i++)
#pragma unroll
        for (int j = 0; j < 2; j++) {
            const int colv = nbase + wn * 32 + j * 16 + lr;
            const float b = bo[colv];
#pragma unroll
            for (int g = 0; g < 8; g++) {
                const int rowv = mbase + wm * 32 + i * 16 + g + half * 8;
                out[(size_t)rowv * OUT_DIM + colv] = acc[i][j][g] + b;
            }
        }
}

// ---------------- Host launch ----------------
extern "C" void kernel_launch(void* const* d_in, const int* in_sizes, int n_in,
                              void* d_out, int out_size, void* d_ws, size_t ws_size,
                              hipStream_t stream) {
    const float* x      = (const float*)d_in[0];
    const float* hidden = (const float*)d_in[1];
    const float* Wxr_w  = (const float*)d_in[2];
    const float* Wxr_b  = (const float*)d_in[3];
    const float* Whr_w  = (const float*)d_in[4];
    const float* Wxz_w  = (const float*)d_in[5];
    const float* Wxz_b  = (const float*)d_in[6];
    const float* Whz_w  = (const float*)d_in[7];
    const float* Wxh_w  = (const float*)d_in[8];
    const float* Wxh_b  = (const float*)d_in[9];
    const float* Whh_w  = (const float*)d_in[10];
    const float* Who_w  = (const float*)d_in[11];
    const float* Who_b  = (const float*)d_in[12];

    float* out_proj = (float*)d_out;                              // (B, OUT)
    float* out_nh   = (float*)d_out + (size_t)BATCH * OUT_DIM;    // (B, HID)

    char* ws = (char*)d_ws;
    size_t off = 0;
    auto carve = [&](size_t bytes) { char* p = ws + off; off += (bytes + 255) & ~(size_t)255; return p; };
    __bf16* xb  = (__bf16*)carve((size_t)BATCH * IN_DIM * 2);
    __bf16* hb  = (__bf16*)carve((size_t)BATCH * HID * 2);
    __bf16* wxr = (__bf16*)carve((size_t)HID * IN_DIM * 2);
    __bf16* wxz = (__bf16*)carve((size_t)HID * IN_DIM * 2);
    __bf16* wxh = (__bf16*)carve((size_t)HID * IN_DIM * 2);
    __bf16* whr = (__bf16*)carve((size_t)HID * HID * 2);
    __bf16* whz = (__bf16*)carve((size_t)HID * HID * 2);
    __bf16* whh = (__bf16*)carve((size_t)HID * HID * 2);
    __bf16* who = (__bf16*)carve((size_t)OUT_DIM * HID * 2);
    __bf16* rhb = (__bf16*)carve((size_t)BATCH * HID * 2);
    float*  zf  = (float*)carve((size_t)BATCH * HID * 4);
    __bf16* nhb = (__bf16*)carve((size_t)BATCH * HID * 2);

    auto cast = [&](const float* s, __bf16* d, long n) {
        int blocks = (int)((n + 255) / 256);
        if (blocks > 4096) blocks = 4096;
        cast_f32_bf16<<<blocks, 256, 0, stream>>>(s, d, n);
    };
    cast(x,      xb,  (long)BATCH * IN_DIM);
    cast(hidden, hb,  (long)BATCH * HID);
    cast(Wxr_w,  wxr, (long)HID * IN_DIM);
    cast(Wxz_w,  wxz, (long)HID * IN_DIM);
    cast(Wxh_w,  wxh, (long)HID * IN_DIM);
    cast(Whr_w,  whr, (long)HID * HID);
    cast(Whz_w,  whz, (long)HID * HID);
    cast(Whh_w,  whh, (long)HID * HID);
    cast(Who_w,  who, (long)OUT_DIM * HID);

    dim3 blk(256);
    dim3 grid_rz(HID / BN, BATCH / BM);     // (16, 128)
    gemm_rz<<<grid_rz, blk, 0, stream>>>(xb, hb, wxr, wxz, whr, whz,
                                         Wxr_b, Wxz_b, hidden, rhb, zf);

    dim3 grid_h(HID / BN, BATCH / BM);      // (16, 128)
    gemm_h<<<grid_h, blk, 0, stream>>>(xb, rhb, wxh, whh, Wxh_b,
                                       zf, hidden, out_nh, nhb);

    dim3 grid_o(OUT_DIM / BN, BATCH / BM);  // (8, 128)
    gemm_out<<<grid_o, blk, 0, stream>>>(nhb, who, Who_b, out_proj);
}
